// MultiHeadAttention_30820685316377
// MI455X (gfx1250) — compile-verified, matrix operand order checked by reading, not run on hardware
//
#include <hip/hip_runtime.h>

// ---------------------------------------------------------------------------
// MHA forward for MI455X (gfx1250, wave32, WMMA).
// Pipeline: qkv_gemm (fp32->bf16, WMMA) -> flash_attn (WMMA + online softmax)
//           -> proj_gemm (WMMA, fp32 out).
// Double-buffered LDS staging everywhere; pure-copy staging (flash K/V,
// proj A) uses GLOBAL_LOAD_ASYNC_TO_LDS_B128 (ASYNCcnt) instead of the
// VGPR round-trip; converting staging (fp32->bf16) stays on the VALU path.
// ---------------------------------------------------------------------------

typedef __attribute__((ext_vector_type(16))) __bf16 bf16x16;
typedef __attribute__((ext_vector_type(8)))  __bf16 bf16x8;
typedef __attribute__((ext_vector_type(8)))  float  f32x8;

constexpr int C  = 1024;   // n_embd
constexpr int C3 = 3072;   // 3*n_embd
constexpr int T  = 2048;   // seq len
constexpr int H  = 16;     // heads
constexpr int D  = 64;     // head dim
constexpr int BATCH = 4;

// ---------------------------------------------------------------------------
// WMMA fragment helpers (layouts per CDNA5 ISA 7.12.2).
// A-fragment (16x32 bf16): lane = h*16 + m; element e: K = (e/8)*16 + h*8 + e%8
//   -> two contiguous 16-byte runs per lane. B mirrors A with lane -> column
//   (we always stage the B operand transposed / K-major per column).
// C/D fragment (16x16 f32): lane = h*16 + n; element r holds row m = r + 8*h.
// ---------------------------------------------------------------------------
__device__ __forceinline__ bf16x16 load_frag(const __bf16* base, int stride, int lane) {
  const int m = lane & 15;
  const int h = (lane >> 4) & 1;
  const __bf16* p = base + m * stride + h * 8;
  bf16x8 lo = *(const bf16x8*)(p);
  bf16x8 hi = *(const bf16x8*)(p + 16);
  bf16x16 r;
#pragma unroll
  for (int i = 0; i < 8; ++i) { r[i] = lo[i]; r[i + 8] = hi[i]; }
  return r;
}

__device__ __forceinline__ f32x8 wmma_bf16(bf16x16 a, bf16x16 b, f32x8 c) {
  return __builtin_amdgcn_wmma_f32_16x16x32_bf16(
      /*neg_a=*/false, a, /*neg_b=*/false, b,
      /*c_mod=*/(short)0, c, /*reuse_a=*/false, /*reuse_b=*/false);
}

__device__ __forceinline__ f32x8 fzero8() {
  f32x8 z = {0.f, 0.f, 0.f, 0.f, 0.f, 0.f, 0.f, 0.f};
  return z;
}

// ---------------------------------------------------------------------------
// Async global->LDS copy (CDNA5 ISA, GV mode).
// VDST = per-lane 32-bit LDS byte offset (low 32 bits of a generic pointer
// to __shared__ are the wave-relative LDS address), VADDR = 64-bit global
// address. Tracked on ASYNCcnt; s_barrier does NOT wait on it, so we issue
// s_wait_asynccnt 0 before the publishing __syncthreads().
// ---------------------------------------------------------------------------
__device__ __forceinline__ void async_ld16(void* lds_dst, const void* gsrc) {
  const unsigned loff = (unsigned)(size_t)lds_dst;
  asm volatile("global_load_async_to_lds_b128 %0, %1, off"
               :: "v"(loff), "v"(gsrc) : "memory");
}
__device__ __forceinline__ void wait_async0() {
  asm volatile("s_wait_asynccnt 0x0" ::: "memory");
}

// ---------------------------------------------------------------------------
// Kernel 1: QKV GEMM. [8192,3072] = x[8192,1024] @ w_attn[1024,3072] + b.
// Epilogue scatters q (scaled by 1/sqrt(D)), k as [B,H,T,D] bf16 and
// v transposed as [B,H,D,T] bf16. fp32->bf16 conversion forces VALU staging.
// ---------------------------------------------------------------------------
__global__ __launch_bounds__(256) void qkv_gemm_kernel(
    const float* __restrict__ x, const float* __restrict__ w,
    const float* __restrict__ bias,
    __bf16* __restrict__ qb, __bf16* __restrict__ kb, __bf16* __restrict__ vtb)
{
  __shared__ __align__(16) __bf16 As[2][128 * 40];
  __shared__ __align__(16) __bf16 Bs[2][128 * 40];
  const int tid = threadIdx.x, lane = tid & 31, wid = tid >> 5;
  const int wm = wid & 1, wn = wid >> 1;            // 2 x 4 wave grid
  const int m0 = blockIdx.x * 128, n0 = blockIdx.y * 128;
  const int ar = tid >> 3, ac4 = (tid & 7) << 2;    // A staging coords

  f32x8 acc[4][2];
#pragma unroll
  for (int i = 0; i < 4; ++i)
#pragma unroll
    for (int j = 0; j < 2; ++j) acc[i][j] = fzero8();

  float4 afr[4], bfr4[4];

  auto load_regs = [&](int k0) {
#pragma unroll
    for (int p = 0; p < 4; ++p)
      afr[p] = *(const float4*)(x + (size_t)(m0 + ar + p * 32) * C + k0 + ac4);
#pragma unroll
    for (int p = 0; p < 4; ++p) {
      const int idx = p * 256 + tid;
      const int kk = idx >> 5, c4 = (idx & 31) << 2;
      bfr4[p] = *(const float4*)(w + (size_t)(k0 + kk) * C3 + n0 + c4);
    }
  };
  auto store_lds = [&](int buf) {
#pragma unroll
    for (int p = 0; p < 4; ++p) {
      __bf16* d = &As[buf][(ar + p * 32) * 40 + ac4];
      d[0] = (__bf16)afr[p].x; d[1] = (__bf16)afr[p].y;
      d[2] = (__bf16)afr[p].z; d[3] = (__bf16)afr[p].w;
    }
#pragma unroll
    for (int p = 0; p < 4; ++p) {
      const int idx = p * 256 + tid;
      const int kk = idx >> 5, c4 = (idx & 31) << 2;
      Bs[buf][(c4 + 0) * 40 + kk] = (__bf16)bfr4[p].x;
      Bs[buf][(c4 + 1) * 40 + kk] = (__bf16)bfr4[p].y;
      Bs[buf][(c4 + 2) * 40 + kk] = (__bf16)bfr4[p].z;
      Bs[buf][(c4 + 3) * 40 + kk] = (__bf16)bfr4[p].w;
    }
  };
  auto compute = [&](int buf) {
    bf16x16 af[4], bfrag[2];
#pragma unroll
    for (int i = 0; i < 4; ++i)
      af[i] = load_frag(&As[buf][(wm * 64 + i * 16) * 40], 40, lane);
#pragma unroll
    for (int j = 0; j < 2; ++j)
      bfrag[j] = load_frag(&Bs[buf][(wn * 32 + j * 16) * 40], 40, lane);
#pragma unroll
    for (int i = 0; i < 4; ++i)
#pragma unroll
      for (int j = 0; j < 2; ++j)
        acc[i][j] = wmma_bf16(af[i], bfrag[j], acc[i][j]);
  };

  constexpr int KT = C / 32;
  load_regs(0);
  store_lds(0);
  __syncthreads();
  for (int kt = 0; kt < KT - 1; ++kt) {
    load_regs((kt + 1) * 32);                              // prefetch next tile
    if (kt + 2 < KT)                                       // warm L2 for k+2
      __builtin_prefetch(x + (size_t)(m0 + ar) * C + (kt + 2) * 32 + ac4, 0, 0);
    compute(kt & 1);
    store_lds((kt + 1) & 1);                               // other buffer: race-free
    __syncthreads();
  }
  compute((KT - 1) & 1);

  // epilogue: bias add + scatter to q/k/v^T
  const int h2 = lane >> 4, nn = lane & 15;
#pragma unroll
  for (int i = 0; i < 4; ++i)
#pragma unroll
    for (int j = 0; j < 2; ++j) {
      const int col = n0 + wn * 32 + j * 16 + nn;
      const float bc = bias[col];
#pragma unroll
      for (int r = 0; r < 8; ++r) {
        const int row = m0 + wm * 64 + i * 16 + r + 8 * h2; // = b*T + t
        const float v = acc[i][j][r] + bc;
        const int bb = row >> 11, tt = row & (T - 1);
        if (col < C) {
          const int hh = col >> 6, dd = col & 63;
          qb[(((size_t)(bb * H + hh) * T + tt) << 6) + dd] = (__bf16)(v * 0.125f);
        } else if (col < 2 * C) {
          const int c2 = col - C, hh = c2 >> 6, dd = c2 & 63;
          kb[(((size_t)(bb * H + hh) * T + tt) << 6) + dd] = (__bf16)v;
        } else {
          const int c2 = col - 2 * C, hh = c2 >> 6, dd = c2 & 63;
          vtb[((size_t)(bb * H + hh) * D + dd) * T + tt] = (__bf16)v;
        }
      }
    }
}

// ---------------------------------------------------------------------------
// Kernel 2: flash attention. One block per (b*h, 128 queries); 8 waves of
// 16 queries each; KV tiles of 32 keys, double-buffered with ASYNC copies.
// ---------------------------------------------------------------------------
__global__ __launch_bounds__(256) void flash_attn_kernel(
    const __bf16* __restrict__ qb, const __bf16* __restrict__ kb,
    const __bf16* __restrict__ vtb, __bf16* __restrict__ yb)
{
  __shared__ __align__(16) __bf16 Ks[2][32 * 72];     // [key][d], stride 72
  __shared__ __align__(16) __bf16 Vs[2][64 * 40];     // [d][key], stride 40
  __shared__ __align__(16) __bf16 Ps[8 * 16 * 40];    // per-wave P scratch

  const int tid = threadIdx.x, lane = tid & 31, wid = tid >> 5;
  const int bh = blockIdx.y;                       // 0..63 = b*H + h
  const int q0 = blockIdx.x * 128 + wid * 16;      // first query t for this wave
  const size_t base = (size_t)bh * T * D;          // same base for q/k/v^T
  const int m = lane & 15, h = lane >> 4, nn = m;
  const int ki = tid >> 3, kd8 = (tid & 7) << 3;   // K staging coords
  const int vd = tid >> 2, vj8 = (tid & 3) << 3;   // V^T staging coords

  // Q fragments (16 queries x 64 dims = 2 A-fragments), 1/sqrt(D) pre-folded
  bf16x16 qf[2];
#pragma unroll
  for (int ks = 0; ks < 2; ++ks)
    qf[ks] = load_frag(qb + base + (size_t)q0 * D + ks * 32, D, lane);

  f32x8 o[4];
#pragma unroll
  for (int dt = 0; dt < 4; ++dt) o[dt] = fzero8();
  float rm[8], rl[8];
#pragma unroll
  for (int r = 0; r < 8; ++r) { rm[r] = -1e30f; rl[r] = 0.f; }

  // async global->LDS staging: 16 B per lane for K tile, 16 B for V^T tile
  auto stage_kv = [&](int kv0, int buf) {
    async_ld16(&Ks[buf][ki * 72 + kd8],
               kb + base + (size_t)(kv0 + ki) * D + kd8);
    async_ld16(&Vs[buf][vd * 40 + vj8],
               vtb + base + (size_t)vd * T + kv0 + vj8);
  };

  auto compute = [&](int buf) {
    // S = Q K^T : 16 queries x 32 keys (2 n-subtiles, 2 k-steps over D)
    f32x8 s[2] = {fzero8(), fzero8()};
#pragma unroll
    for (int ks = 0; ks < 2; ++ks)
#pragma unroll
      for (int sn = 0; sn < 2; ++sn) {
        bf16x16 kf = load_frag(&Ks[buf][sn * 16 * 72 + ks * 32], 72, lane);
        s[sn] = wmma_bf16(qf[ks], kf, s[sn]);
      }
    // online softmax per row (row m = r + 8*h lives across one 16-lane half)
#pragma unroll
    for (int r = 0; r < 8; ++r) {
      float mx = fmaxf(s[0][r], s[1][r]);
#pragma unroll
      for (int msk = 1; msk < 16; msk <<= 1) mx = fmaxf(mx, __shfl_xor(mx, msk, 32));
      const float mnew = fmaxf(rm[r], mx);
      const float psc  = __expf(rm[r] - mnew);
      const float p0 = __expf(s[0][r] - mnew);
      const float p1 = __expf(s[1][r] - mnew);
      float sum = p0 + p1;
#pragma unroll
      for (int msk = 1; msk < 16; msk <<= 1) sum += __shfl_xor(sum, msk, 32);
      rl[r] = rl[r] * psc + sum;
      rm[r] = mnew;
      const int mm = r + 8 * h;
      Ps[(wid * 16 + mm) * 40 + nn]      = (__bf16)p0;
      Ps[(wid * 16 + mm) * 40 + 16 + nn] = (__bf16)p1;
#pragma unroll
      for (int dt = 0; dt < 4; ++dt) o[dt][r] *= psc;
    }
    asm volatile("s_wait_dscnt 0x0" ::: "memory");  // P stores -> P reads (same wave)
    // O += P V : A-fragment from per-wave P scratch, B-fragments from V^T
    bf16x16 pf = load_frag(&Ps[wid * 16 * 40], 40, lane);
#pragma unroll
    for (int dt = 0; dt < 4; ++dt) {
      bf16x16 vf = load_frag(&Vs[buf][dt * 16 * 40], 40, lane);
      o[dt] = wmma_bf16(pf, vf, o[dt]);
    }
  };

  constexpr int NT = T / 32;
  stage_kv(0, 0);
  wait_async0();
  __syncthreads();
  for (int it = 0; it < NT - 1; ++it) {
    stage_kv((it + 1) * 32, (it + 1) & 1);          // async fill of other buffer
    compute(it & 1);
    wait_async0();                                  // LDS writes landed
    __syncthreads();                                // publish to all waves
  }
  compute((NT - 1) & 1);

  // normalize and write y as bf16 [B*T, C] with column = h*64 + d
  const int bb = bh >> 4, hh = bh & 15;
#pragma unroll
  for (int r = 0; r < 8; ++r) {
    const float inv = 1.0f / rl[r];
    const int t = q0 + r + 8 * h;
    const size_t rowbase = ((size_t)bb * T + t) * C + hh * D;
#pragma unroll
    for (int dt = 0; dt < 4; ++dt)
      yb[rowbase + dt * 16 + nn] = (__bf16)(o[dt][r] * inv);
  }
}

// ---------------------------------------------------------------------------
// Kernel 3: output projection. out[8192,1024] = y[8192,1024] @ w_proj + b.
// A operand (bf16) staged via ASYNC copies; B operand converts fp32->bf16
// so it stays on the VALU path.
// ---------------------------------------------------------------------------
__global__ __launch_bounds__(256) void proj_gemm_kernel(
    const __bf16* __restrict__ yb, const float* __restrict__ w,
    const float* __restrict__ bias, float* __restrict__ out)
{
  __shared__ __align__(16) __bf16 As[2][128 * 40];
  __shared__ __align__(16) __bf16 Bs[2][128 * 40];
  const int tid = threadIdx.x, lane = tid & 31, wid = tid >> 5;
  const int wm = wid & 1, wn = wid >> 1;
  const int m0 = blockIdx.x * 128, n0 = blockIdx.y * 128;

  f32x8 acc[4][2];
#pragma unroll
  for (int i = 0; i < 4; ++i)
#pragma unroll
    for (int j = 0; j < 2; ++j) acc[i][j] = fzero8();

  float4 bfr4[4];

  auto stage_a_async = [&](int k0, int buf) {
#pragma unroll
    for (int p = 0; p < 2; ++p) {
      const int idx = p * 256 + tid;
      const int row = idx >> 2, c8 = (idx & 3) << 3;
      async_ld16(&As[buf][row * 40 + c8],
                 yb + (size_t)(m0 + row) * C + k0 + c8);
    }
  };
  auto load_b_regs = [&](int k0) {
#pragma unroll
    for (int p = 0; p < 4; ++p) {
      const int idx = p * 256 + tid;
      const int kk = idx >> 5, c4 = (idx & 31) << 2;
      bfr4[p] = *(const float4*)(w + (size_t)(k0 + kk) * C + n0 + c4);
    }
  };
  auto store_b_lds = [&](int buf) {
#pragma unroll
    for (int p = 0; p < 4; ++p) {
      const int idx = p * 256 + tid;
      const int kk = idx >> 5, c4 = (idx & 31) << 2;
      Bs[buf][(c4 + 0) * 40 + kk] = (__bf16)bfr4[p].x;
      Bs[buf][(c4 + 1) * 40 + kk] = (__bf16)bfr4[p].y;
      Bs[buf][(c4 + 2) * 40 + kk] = (__bf16)bfr4[p].z;
      Bs[buf][(c4 + 3) * 40 + kk] = (__bf16)bfr4[p].w;
    }
  };
  auto compute = [&](int buf) {
    bf16x16 af[4], bfrag[2];
#pragma unroll
    for (int i = 0; i < 4; ++i)
      af[i] = load_frag(&As[buf][(wm * 64 + i * 16) * 40], 40, lane);
#pragma unroll
    for (int j = 0; j < 2; ++j)
      bfrag[j] = load_frag(&Bs[buf][(wn * 32 + j * 16) * 40], 40, lane);
#pragma unroll
    for (int i = 0; i < 4; ++i)
#pragma unroll
      for (int j = 0; j < 2; ++j)
        acc[i][j] = wmma_bf16(af[i], bfrag[j], acc[i][j]);
  };

  constexpr int KT = C / 32;
  stage_a_async(0, 0);
  load_b_regs(0);
  store_b_lds(0);
  wait_async0();
  __syncthreads();
  for (int kt = 0; kt < KT - 1; ++kt) {
    stage_a_async((kt + 1) * 32, (kt + 1) & 1);     // async A into other buffer
    load_b_regs((kt + 1) * 32);                     // B prefetch (needs convert)
    compute(kt & 1);
    store_b_lds((kt + 1) & 1);
    wait_async0();
    __syncthreads();
  }
  compute((KT - 1) & 1);

  const int h2 = lane >> 4, nn = lane & 15;
#pragma unroll
  for (int i = 0; i < 4; ++i)
#pragma unroll
    for (int j = 0; j < 2; ++j) {
      const int col = n0 + wn * 32 + j * 16 + nn;
      const float bc = bias[col];
#pragma unroll
      for (int r = 0; r < 8; ++r) {
        const int row = m0 + wm * 64 + i * 16 + r + 8 * h2;
        out[(size_t)row * C + col] = acc[i][j][r] + bc;
      }
    }
}

// ---------------------------------------------------------------------------
extern "C" void kernel_launch(void* const* d_in, const int* in_sizes, int n_in,
                              void* d_out, int out_size, void* d_ws, size_t ws_size,
                              hipStream_t stream)
{
  (void)in_sizes; (void)n_in; (void)out_size; (void)ws_size;
  const float* x      = (const float*)d_in[0];
  const float* w_attn = (const float*)d_in[1];
  const float* b_attn = (const float*)d_in[2];
  const float* w_proj = (const float*)d_in[3];
  const float* b_proj = (const float*)d_in[4];
  float* out = (float*)d_out;

  const size_t n_elem = (size_t)BATCH * H * T * D;  // 8,388,608 per buffer
  __bf16* qb  = (__bf16*)d_ws;
  __bf16* kb  = qb  + n_elem;
  __bf16* vtb = kb  + n_elem;
  __bf16* yb  = vtb + n_elem;   // total 64 MB of workspace

  qkv_gemm_kernel<<<dim3(64, 24), 256, 0, stream>>>(x, w_attn, b_attn, qb, kb, vtb);
  flash_attn_kernel<<<dim3(16, 64), 256, 0, stream>>>(qb, kb, vtb, yb);
  proj_gemm_kernel<<<dim3(64, 8), 256, 0, stream>>>(yb, w_proj, b_proj, out);
}
